// MultiHeadAttention_70720931496749
// MI455X (gfx1250) — matrix operand order repaired and checked by reading, not compiled, not run on hardware
//
#include <hip/hip_runtime.h>
#include <hip/hip_bf16.h>

// ---------------------------------------------------------------------------
// MI455X (gfx1250) multi-head attention, bf16 WMMA pipeline, f32 accumulate.
// Stages: f32->bf16 convert | QKV projection GEMMs (V stored transposed)
//         | flash-attention (online softmax) + residual.
// ---------------------------------------------------------------------------

typedef __bf16 bf16;
typedef bf16  v8bf  __attribute__((ext_vector_type(8)));
typedef bf16  v16bf __attribute__((ext_vector_type(16)));
typedef float v8f   __attribute__((ext_vector_type(8)));

#define B_  4
#define L_  2048
#define D_  512
#define H_  8
#define DH_ 64

static __device__ __forceinline__ v8f wmma_bf16(v16bf a, v16bf b, v8f c) {
  // (neg_a, A, neg_b, B, c_mod, C, reuse_a, reuse_b)
  return __builtin_amdgcn_wmma_f32_16x16x32_bf16(false, a, false, b, (short)0, c,
                                                 false, false);
}

// A-operand (16x32 bf16): lane holds row (lane&15); K = {kb..kb+15}
// with kb = 16*(lane>=16). Caller passes row pointer already offset by kb.
static __device__ __forceinline__ v16bf ld_a_frag(const bf16* p) {
  return *(const v16bf*)p;
}

// B-operand (32x16 bf16): lane = column N; 16 consecutive K values starting at
// 16*(lane>=16). Caller passes the B^T-row pointer already offset by hi*16.
static __device__ __forceinline__ v16bf ld_b_frag(const bf16* p) {
  return *(const v16bf*)p;
}

// ---------------------------------------------------------------------------
__global__ void cvt_bf16_kernel(const float* __restrict__ in,
                                bf16* __restrict__ out, int n) {
  int i = blockIdx.x * blockDim.x + threadIdx.x;
  if (i < n) out[i] = (bf16)in[i];
}

// ---------------------------------------------------------------------------
// Y[m][n] = sum_k X[m][k] * W[n][k]   (torch nn.Linear: x @ W^T)
// Each wave owns a 16(M) x 64(N) tile; block = 8 waves -> 128 rows.
// vmode==0: Y row-major bf16 (M x D).  vmode==1: store V transposed per head:
//           Vt[((b*H+h)*DH + d) * L + l].
__global__ __launch_bounds__(256)
void qkv_gemm_kernel(const bf16* __restrict__ X, const bf16* __restrict__ W,
                     bf16* __restrict__ Y, int vmode) {
  const int lane = threadIdx.x & 31;
  const int wave = threadIdx.x >> 5;
  const int lm = lane & 15;
  const int hi = lane >> 4;
  const int row0 = (blockIdx.x * 8 + wave) * 16;
  const int n0 = blockIdx.y * 64;

  const bf16* arow = X + (size_t)(row0 + lm) * D_;
  v8f acc[4] = {};

  for (int k = 0; k < D_; k += 32) {
    v16bf a = ld_a_frag(arow + k + hi * 16);
#pragma unroll
    for (int j = 0; j < 4; ++j) {
      v16bf bb = ld_b_frag(W + (size_t)(n0 + j * 16 + lm) * D_ + k + hi * 16);
      acc[j] = wmma_bf16(a, bb, acc[j]);
    }
  }

#pragma unroll
  for (int j = 0; j < 4; ++j) {
#pragma unroll
    for (int r = 0; r < 8; ++r) {
      int m = row0 + r + hi * 8;           // C layout: lanes>=16 own rows r+8
      int n = n0 + j * 16 + lm;
      bf16 v = (bf16)acc[j][r];
      if (vmode == 0) {
        Y[(size_t)m * D_ + n] = v;
      } else {
        int bb = m >> 11, l = m & (L_ - 1);
        int hh = n >> 6, d = n & (DH_ - 1);
        Y[((size_t)((bb * H_ + hh) * DH_ + d)) * L_ + l] = v;
      }
    }
  }
}

// ---------------------------------------------------------------------------
// Flash attention: one wave per 16-query block; iterate keys in chunks of 32.
// Per chunk: 4 WMMA for scores (two 16-key score tiles, dh=64 = 2x K32),
// online softmax with shfl_xor row reductions, P re-swizzled via LDS into the
// A layout, 4 WMMA for P.V into the 16x64 output accumulator.
__global__ __launch_bounds__(256)
void attn_kernel(const bf16* __restrict__ Qb, const bf16* __restrict__ Kb,
                 const bf16* __restrict__ Vt, const float* __restrict__ query,
                 float* __restrict__ out) {
  __shared__ bf16 lds_p[8][16 * 32];   // per-wave 16x32 P staging tile (1 KB)

  const int lane = threadIdx.x & 31;
  const int wave = threadIdx.x >> 5;
  const int lm = lane & 15;
  const int hi = lane >> 4;
  const int kb = hi * 16;
  const int q0 = (blockIdx.x * 8 + wave) * 16;
  const int h = blockIdx.y;
  const int b = blockIdx.z;
  const float inv_scale = 0.04419417382415922f;  // 1/sqrt(D), per reference

  const bf16* qrow = Qb + (size_t)(b * L_ + q0 + lm) * D_ + h * DH_;
  v16bf aq0 = ld_a_frag(qrow + kb);        // dk 0..31
  v16bf aq1 = ld_a_frag(qrow + 32 + kb);   // dk 32..63

  float rmax[8], rsum[8];
#pragma unroll
  for (int r = 0; r < 8; ++r) { rmax[r] = -1e30f; rsum[r] = 0.f; }
  v8f o[4] = {};

  bf16* lp = lds_p[wave];
  const bf16* vhead = Vt + (size_t)((b * H_ + h) * DH_) * L_;

  for (int k0 = 0; k0 < L_; k0 += 32) {
    const bf16* kr0 = Kb + (size_t)(b * L_ + k0 + lm) * D_ + h * DH_;
    const bf16* kr1 = kr0 + (size_t)16 * D_;
    v16bf bk0a = ld_b_frag(kr0 + hi * 16);        // keys k0..15, dk 0..31
    v16bf bk0b = ld_b_frag(kr0 + 32 + hi * 16);   // keys k0..15, dk 32..63
    v16bf bk1a = ld_b_frag(kr1 + hi * 16);        // keys k0+16..31
    v16bf bk1b = ld_b_frag(kr1 + 32 + hi * 16);

    v8f s0 = {}, s1 = {};
    s0 = wmma_bf16(aq0, bk0a, s0);
    s0 = wmma_bf16(aq1, bk0b, s0);
    s1 = wmma_bf16(aq0, bk1a, s1);
    s1 = wmma_bf16(aq1, bk1b, s1);

#pragma unroll
    for (int r = 0; r < 8; ++r) {
      float a0 = s0[r] * inv_scale;
      float a1 = s1[r] * inv_scale;
      // row max across the 16 lanes of this half (rows map: M=r / M=r+8)
      float t = fmaxf(a0, a1);
      t = fmaxf(t, __shfl_xor(t, 1, 32));
      t = fmaxf(t, __shfl_xor(t, 2, 32));
      t = fmaxf(t, __shfl_xor(t, 4, 32));
      t = fmaxf(t, __shfl_xor(t, 8, 32));
      float nm = fmaxf(rmax[r], t);
      float corr = __expf(rmax[r] - nm);
      rmax[r] = nm;
      float p0 = __expf(a0 - nm);
      float p1 = __expf(a1 - nm);
      float ts = p0 + p1;
      ts += __shfl_xor(ts, 1, 32);
      ts += __shfl_xor(ts, 2, 32);
      ts += __shfl_xor(ts, 4, 32);
      ts += __shfl_xor(ts, 8, 32);
      rsum[r] = rsum[r] * corr + ts;
      o[0][r] *= corr; o[1][r] *= corr; o[2][r] *= corr; o[3][r] *= corr;
      // stage P into LDS row-major (16 rows x 32 key-cols), C->A relayout
      int prow = r + hi * 8;
      lp[prow * 32 + lm] = (bf16)p0;
      lp[prow * 32 + 16 + lm] = (bf16)p1;
    }
    // Same-wave DS ops are in-order on CDNA5; only block compiler reordering.
    asm volatile("" ::: "memory");
    v16bf pf = ld_a_frag(lp + lm * 32 + kb);
    asm volatile("" ::: "memory");

#pragma unroll
    for (int j = 0; j < 4; ++j) {
      const bf16* vr = vhead + (size_t)(j * 16 + lm) * L_ + k0 + hi * 16;
      v16bf bv = ld_b_frag(vr);
      o[j] = wmma_bf16(pf, bv, o[j]);
    }
  }

  // epilogue: normalize, residual add, store f32
#pragma unroll
  for (int j = 0; j < 4; ++j) {
#pragma unroll
    for (int r = 0; r < 8; ++r) {
      int qg = q0 + r + hi * 8;
      int d = j * 16 + lm;
      size_t idx = (size_t)(b * L_ + qg) * D_ + h * DH_ + d;
      out[idx] = o[j][r] / rsum[r] + query[idx];
    }
  }
}

// ---------------------------------------------------------------------------
extern "C" void kernel_launch(void* const* d_in, const int* in_sizes, int n_in,
                              void* d_out, int out_size, void* d_ws, size_t ws_size,
                              hipStream_t stream) {
  (void)in_sizes; (void)n_in; (void)out_size; (void)ws_size;
  const float* query = (const float*)d_in[0];
  const float* keys  = (const float*)d_in[1];
  const float* Wq    = (const float*)d_in[2];
  const float* Wk    = (const float*)d_in[3];
  const float* Wv    = (const float*)d_in[4];
  float* out = (float*)d_out;

  char* ws = (char*)d_ws;
  const size_t sz_x = (size_t)B_ * L_ * D_ * sizeof(bf16);  // 8 MB
  const size_t sz_w = (size_t)D_ * D_ * sizeof(bf16);       // 0.5 MB
  bf16* Xq  = (bf16*)ws; ws += sz_x;
  bf16* Xk  = (bf16*)ws; ws += sz_x;
  bf16* Wqb = (bf16*)ws; ws += sz_w;
  bf16* Wkb = (bf16*)ws; ws += sz_w;
  bf16* Wvb = (bf16*)ws; ws += sz_w;
  bf16* Qb  = (bf16*)ws; ws += sz_x;
  bf16* Kb  = (bf16*)ws; ws += sz_x;
  bf16* Vt  = (bf16*)ws; ws += sz_x;   // total ~41.5 MB workspace

  const int nx = B_ * L_ * D_;
  const int nw = D_ * D_;
  cvt_bf16_kernel<<<(nx + 255) / 256, 256, 0, stream>>>(query, Xq, nx);
  cvt_bf16_kernel<<<(nx + 255) / 256, 256, 0, stream>>>(keys,  Xk, nx);
  cvt_bf16_kernel<<<(nw + 255) / 256, 256, 0, stream>>>(Wq, Wqb, nw);
  cvt_bf16_kernel<<<(nw + 255) / 256, 256, 0, stream>>>(Wk, Wkb, nw);
  cvt_bf16_kernel<<<(nw + 255) / 256, 256, 0, stream>>>(Wv, Wvb, nw);

  dim3 gg(B_ * L_ / 128, D_ / 64);  // 64 x 8 blocks, 8 waves/block
  qkv_gemm_kernel<<<gg, 256, 0, stream>>>(Xq, Wqb, Qb, 0);
  qkv_gemm_kernel<<<gg, 256, 0, stream>>>(Xk, Wkb, Kb, 0);
  qkv_gemm_kernel<<<gg, 256, 0, stream>>>(Xk, Wvb, Vt, 1);

  dim3 ga(L_ / 128, H_, B_);        // 16 x 8 x 4 blocks, 8 waves/block
  attn_kernel<<<ga, 256, 0, stream>>>(Qb, Kb, Vt, query, out);
}
